// Attention_8211977470693
// MI455X (gfx1250) — matrix operand order repaired and checked by reading, not compiled, not run on hardware
//
#include <hip/hip_runtime.h>

// MI455X / gfx1250, wave32. Flash-attention MHA in 3 WMMA-bf16 kernels.
// Workspace (bf16 elements): Qb | Kb | Vb | Obuf, 8192*1024 each (64MB total).

constexpr int kB     = 4;
constexpr int kN     = 2048;
constexpr int kDIM   = 1024;
constexpr int kH     = 16;
constexpr int kDH    = 64;
constexpr int kINNER = kH * kDH;   // 1024
constexpr int kM     = kB * kN;    // 8192
constexpr int kNQKV  = 3 * kINNER; // 3072

typedef __attribute__((ext_vector_type(16))) __bf16 v16bf;
typedef __attribute__((ext_vector_type(8)))  float  v8f;

union Frag {
    v16bf v;
    unsigned int w[8];
};

__device__ __forceinline__ unsigned short f2bf(float f) {
    unsigned int u = __float_as_uint(f);
    unsigned int r = (u + 0x7FFFu + ((u >> 16) & 1u)) >> 16;  // RNE
    return (unsigned short)r;
}

__device__ __forceinline__ v8f wmma_bf16(const Frag& a, const Frag& b, v8f c) {
    return __builtin_amdgcn_wmma_f32_16x16x32_bf16(
        false, a.v, false, b.v, (short)0, c, false, false);
}

// A-fragment K index (16-bit A 16x32): 16*half + 2j (contiguous, matches B)
__device__ __forceinline__ int a_kidx(int j, int half) {
    return 16 * half + 2 * j;
}

// ---- CDNA5 async DMA: global -> LDS, 128-bit, tracked by ASYNCcnt ----------
__device__ __forceinline__ void async_ld_b128(const unsigned short* gsrc,
                                              const unsigned short* ldst) {
    unsigned loff = (unsigned)(unsigned long long)ldst;  // LDS window offset
    asm volatile("global_load_async_to_lds_b128 %0, %1, off"
                 :: "v"(loff), "v"(gsrc) : "memory");
}
__device__ __forceinline__ void wait_async0() {
    asm volatile("s_wait_asynccnt 0x0" ::: "memory");
}

// ---------------------------------------------------------------------------
// Kernel 1: QKV projection + RoPE. C[8192 x 3072] = x @ [Wq|Wkv], bf16 WMMA.
// 128x128 tile per 256-thread block (8 waves x 16 rows x 128 cols), K-stage 64.
// ---------------------------------------------------------------------------
__global__ __launch_bounds__(256)
void qkv_rope_kernel(const float* __restrict__ x,
                     const float* __restrict__ rope,   // [N, DH]
                     const float* __restrict__ Wq,     // [1024,1024]
                     const float* __restrict__ Wkv,    // [1024,2048]
                     unsigned short* __restrict__ Qb,  // [B,H,N,DH] bf16
                     unsigned short* __restrict__ Kb,
                     unsigned short* __restrict__ Vb) {
    __shared__ unsigned short As[128 * 64];   // [row][k]
    __shared__ unsigned short Bs[128 * 64];   // [col][k]

    const int t = threadIdx.x;
    const int wv = t >> 5, lane = t & 31;
    const int half = lane >> 4, l16 = lane & 15;
    const int row0 = blockIdx.y * 128;
    const int col0 = blockIdx.x * 128;

    // uniform region selection (128-col stripe never straddles q/k/v boundary)
    const float* Wsrc; size_t wstride; int wcol0;
    if (col0 < kINNER)      { Wsrc = Wq;  wstride = kINNER;     wcol0 = col0; }
    else                    { Wsrc = Wkv; wstride = 2 * kINNER; wcol0 = col0 - kINNER; }

    v8f acc[8];
    #pragma unroll
    for (int ct = 0; ct < 8; ++ct)
        #pragma unroll
        for (int e = 0; e < 8; ++e) acc[ct][e] = 0.f;

    for (int k0 = 0; k0 < kDIM; k0 += 64) {
        __syncthreads();
        for (int i = 0; i < 32; ++i) {                 // A: 128x64 f32 -> bf16
            int idx = i * 256 + t;
            int r = idx >> 6, k = idx & 63;
            As[idx] = f2bf(x[(size_t)(row0 + r) * kDIM + k0 + k]);
        }
        for (int i = 0; i < 32; ++i) {                 // B transposed: Bs[col][k]
            int idx = i * 256 + t;
            int k = idx >> 7, c = idx & 127;
            Bs[c * 64 + k] = f2bf(Wsrc[(size_t)(k0 + k) * wstride + wcol0 + c]);
        }
        __syncthreads();
        #pragma unroll
        for (int c = 0; c < 2; ++c) {
            Frag af;
            #pragma unroll
            for (int j = 0; j < 8; ++j)
                af.w[j] = *(const unsigned int*)
                    &As[(wv * 16 + l16) * 64 + c * 32 + a_kidx(j, half)];
            #pragma unroll
            for (int ct = 0; ct < 8; ++ct) {
                Frag bf;
                #pragma unroll
                for (int j = 0; j < 8; ++j)
                    bf.w[j] = *(const unsigned int*)
                        &Bs[(ct * 16 + l16) * 64 + c * 32 + half * 16 + 2 * j];
                acc[ct] = wmma_bf16(af, bf, acc[ct]);
            }
        }
    }

    // epilogue: constant-indexed copy, scalar region branch, RoPE in-register
    float a[8][8];
    #pragma unroll
    for (int ct = 0; ct < 8; ++ct)
        #pragma unroll
        for (int e = 0; e < 8; ++e) a[ct][e] = acc[ct][e];

    const int region = col0 >> 10;   // 0=q, 1=k, 2=v (uniform)
    if (region == 2) {
        #pragma unroll
        for (int ct = 0; ct < 8; ++ct) {
            int nc = (col0 & 1023) + ct * 16 + l16;
            int h = nc >> 6, d = nc & 63;
            #pragma unroll
            for (int i = 0; i < 8; ++i) {
                int r = row0 + wv * 16 + i + 8 * half;
                int b = r >> 11, seq = r & (kN - 1);
                Vb[((size_t)(b * kH + h) * kN + seq) * kDH + d] = f2bf(a[ct][i]);
            }
        }
    } else {
        unsigned short* dst = (region == 0) ? Qb : Kb;
        const float sc = (region == 0) ? 0.125f : 1.0f;  // fold softmax scale into q
        #pragma unroll
        for (int ct = 0; ct < 8; ++ct) {
            int nc = (col0 & 1023) + ct * 16 + l16;
            int h = nc >> 6, d = nc & 63;
            float sgn = (d < 32) ? -1.f : 1.f;           // rotate_half sign
            #pragma unroll
            for (int i = 0; i < 8; ++i) {
                int r = row0 + wv * 16 + i + 8 * half;
                int b = r >> 11, seq = r & (kN - 1);
                float pos = rope[seq * kDH + d];
                float val = (a[ct][i] * __cosf(pos) +
                             sgn * a[ct ^ 2][i] * __sinf(pos)) * sc;
                dst[((size_t)(b * kH + h) * kN + seq) * kDH + d] = f2bf(val);
            }
        }
    }
}

// ---------------------------------------------------------------------------
// Kernel 2: flash attention. Block = (b,h, 128 q-rows), 8 waves x 16 rows.
// K/V staged via async global->LDS DMA (ASYNCcnt), V transposed in LDS.
// ---------------------------------------------------------------------------
__global__ __launch_bounds__(256)
void attn_kernel(const unsigned short* __restrict__ Qb,
                 const unsigned short* __restrict__ Kb,
                 const unsigned short* __restrict__ Vb,
                 unsigned short* __restrict__ Obuf) {  // [B,N,H*DH] bf16
    __shared__ unsigned short Ks[64 * 64];      // [key][d]
    __shared__ unsigned short Vrow[64 * 64];    // [key][d] async landing
    __shared__ unsigned short VsT[64 * 64];     // [d][key]
    __shared__ float Ps[8 * 16 * 64];           // per-wave scores (f32)
    __shared__ unsigned short Pw[8 * 16 * 64];  // per-wave P (bf16, A-layout feed)
    __shared__ float crr[8 * 16];

    const int t = threadIdx.x;
    const int wv = t >> 5, lane = t & 31;
    const int half = lane >> 4, l16 = lane & 15;
    const int bh = blockIdx.y;
    const int b = bh >> 4, h = bh & 15;
    const int q0 = blockIdx.x * 128;
    const size_t base = (size_t)bh * kN * kDH;

    const int qi = q0 + wv * 16 + l16;
    Frag qf[2];
    #pragma unroll
    for (int c = 0; c < 2; ++c)
        #pragma unroll
        for (int j = 0; j < 8; ++j) {
            int dd = c * 32 + a_kidx(j, half);
            qf[c].w[j] = *(const unsigned int*)&Qb[base + (size_t)qi * kDH + dd];
        }

    v8f o[4];
    #pragma unroll
    for (int ot = 0; ot < 4; ++ot)
        #pragma unroll
        for (int e = 0; e < 8; ++e) o[ot][e] = 0.f;
    float m_run = -1e30f, l_run = 0.f;

    for (int key0 = 0; key0 < kN; key0 += 64) {
        __syncthreads();
        // async stage: K and V blocks, 64x64 bf16 each = 512 b128 transfers each
        #pragma unroll
        for (int i = 0; i < 2; ++i) {
            int eb = i * 256 + t;
            int key = eb >> 3, seg = eb & 7;
            const unsigned short* kg = Kb + base + (size_t)(key0 + key) * kDH + seg * 8;
            const unsigned short* vg = Vb + base + (size_t)(key0 + key) * kDH + seg * 8;
            async_ld_b128(kg, &Ks[key * 64 + seg * 8]);
            async_ld_b128(vg, &Vrow[key * 64 + seg * 8]);
        }
        wait_async0();
        __syncthreads();
        // transpose V in LDS: VsT[d][key]
        for (int i = 0; i < 16; ++i) {
            int e = i * 256 + t;
            int d = e >> 6, key = e & 63;
            VsT[d * 64 + key] = Vrow[key * 64 + d];
        }
        __syncthreads();

        // S = q @ K^T (scale folded into q): 4 key-tiles x 2 d-chunks
        float* ps = &Ps[wv * 16 * 64];
        #pragma unroll
        for (int ct = 0; ct < 4; ++ct) {
            v8f acc;
            #pragma unroll
            for (int e = 0; e < 8; ++e) acc[e] = 0.f;
            int ck = ct * 16 + l16;
            #pragma unroll
            for (int c = 0; c < 2; ++c) {
                Frag kf;
                #pragma unroll
                for (int j = 0; j < 8; ++j) {
                    int dd = c * 32 + half * 16 + 2 * j;
                    kf.w[j] = *(const unsigned int*)&Ks[ck * 64 + dd];
                }
                acc = wmma_bf16(qf[c], kf, acc);
            }
            #pragma unroll
            for (int i = 0; i < 8; ++i)
                ps[(i + 8 * half) * 64 + ct * 16 + l16] = acc[i];
        }

        // online softmax: lanes 0..15 own one row each (wave-local)
        unsigned short* pw = &Pw[wv * 16 * 64];
        if (lane < 16) {
            const float* prow = &ps[lane * 64];
            float bm = -1e30f;
            for (int c2 = 0; c2 < 64; ++c2) bm = fmaxf(bm, prow[c2]);
            float mnew = fmaxf(m_run, bm);
            float cr = __expf(m_run - mnew);
            float s = 0.f;
            for (int c2 = 0; c2 < 64; ++c2) {
                float e_ = __expf(prow[c2] - mnew);
                s += e_;
                pw[lane * 64 + c2] = f2bf(e_);
            }
            l_run = l_run * cr + s;
            m_run = mnew;
            crr[wv * 16 + lane] = cr;
        }

        // rescale running output
        #pragma unroll
        for (int ot = 0; ot < 4; ++ot)
            #pragma unroll
            for (int i = 0; i < 8; ++i)
                o[ot][i] *= crr[wv * 16 + i + 8 * half];

        // O += P @ V
        Frag pf[2];
        #pragma unroll
        for (int c = 0; c < 2; ++c)
            #pragma unroll
            for (int j = 0; j < 8; ++j) {
                int kk = c * 32 + a_kidx(j, half);
                pf[c].w[j] = *(const unsigned int*)&pw[l16 * 64 + kk];
            }
        #pragma unroll
        for (int ot = 0; ot < 4; ++ot) {
            int d = ot * 16 + l16;
            #pragma unroll
            for (int c = 0; c < 2; ++c) {
                Frag vf;
                #pragma unroll
                for (int j = 0; j < 8; ++j) {
                    int kk = c * 32 + half * 16 + 2 * j;
                    vf.w[j] = *(const unsigned int*)&VsT[d * 64 + kk];
                }
                o[ot] = wmma_bf16(pf[c], vf, o[ot]);
            }
        }
    }

    // final 1/l normalize + store bf16 into [B, N, H*DH]
    if (lane < 16) crr[wv * 16 + lane] = 1.f / l_run;
    #pragma unroll
    for (int ot = 0; ot < 4; ++ot) {
        int dcol = h * kDH + ot * 16 + l16;
        #pragma unroll
        for (int i = 0; i < 8; ++i) {
            int qg = q0 + wv * 16 + i + 8 * half;
            float val = o[ot][i] * crr[wv * 16 + i + 8 * half];
            Obuf[(size_t)(b * kN + qg) * kINNER + dcol] = f2bf(val);
        }
    }
}

// ---------------------------------------------------------------------------
// Kernel 3: output projection. out[8192x1024] = Obuf(bf16) @ Wo + bo (f32 out).
// 128x128 tile / 256 threads; A staged via async DMA, B converted on the fly.
// ---------------------------------------------------------------------------
__global__ __launch_bounds__(256)
void oproj_kernel(const unsigned short* __restrict__ Obuf,
                  const float* __restrict__ Wo,
                  const float* __restrict__ bo,
                  float* __restrict__ out) {
    __shared__ unsigned short As[128 * 64];   // [row][k]
    __shared__ unsigned short Bs[128 * 64];   // [col][k]
    const int t = threadIdx.x;
    const int wv = t >> 5, lane = t & 31;
    const int half = lane >> 4, l16 = lane & 15;
    const int row0 = blockIdx.y * 128;
    const int col0 = blockIdx.x * 128;

    v8f acc[8];
    #pragma unroll
    for (int ct = 0; ct < 8; ++ct)
        #pragma unroll
        for (int e = 0; e < 8; ++e) acc[ct][e] = 0.f;

    for (int k0 = 0; k0 < kINNER; k0 += 64) {
        __syncthreads();
        // A: async DMA, 128x64 bf16 = 1024 b128 transfers
        #pragma unroll
        for (int i = 0; i < 4; ++i) {
            int eb = i * 256 + t;
            int r = eb >> 3, seg = eb & 7;
            async_ld_b128(Obuf + (size_t)(row0 + r) * kINNER + k0 + seg * 8,
                          &As[r * 64 + seg * 8]);
        }
        // B: f32 -> bf16, transposed Bs[col][k] (overlaps with async A)
        for (int i = 0; i < 32; ++i) {
            int idx = i * 256 + t;
            int k = idx >> 7, c = idx & 127;
            Bs[c * 64 + k] = f2bf(Wo[(size_t)(k0 + k) * kDIM + col0 + c]);
        }
        wait_async0();
        __syncthreads();
        #pragma unroll
        for (int c = 0; c < 2; ++c) {
            Frag af;
            #pragma unroll
            for (int j = 0; j < 8; ++j)
                af.w[j] = *(const unsigned int*)
                    &As[(wv * 16 + l16) * 64 + c * 32 + a_kidx(j, half)];
            #pragma unroll
            for (int ct = 0; ct < 8; ++ct) {
                Frag bf;
                #pragma unroll
                for (int j = 0; j < 8; ++j)
                    bf.w[j] = *(const unsigned int*)
                        &Bs[(ct * 16 + l16) * 64 + c * 32 + half * 16 + 2 * j];
                acc[ct] = wmma_bf16(af, bf, acc[ct]);
            }
        }
    }

    #pragma unroll
    for (int ct = 0; ct < 8; ++ct) {
        int n = col0 + ct * 16 + l16;
        float bias = bo[n];
        #pragma unroll
        for (int i = 0; i < 8; ++i) {
            int r = row0 + wv * 16 + i + 8 * half;
            out[(size_t)r * kDIM + n] = acc[ct][i] + bias;
        }
    }
}

extern "C" void kernel_launch(void* const* d_in, const int* in_sizes, int n_in,
                              void* d_out, int out_size, void* d_ws, size_t ws_size,
                              hipStream_t stream) {
    (void)in_sizes; (void)n_in; (void)out_size; (void)ws_size;
    const float* x    = (const float*)d_in[0];
    const float* rope = (const float*)d_in[1];
    const float* Wq   = (const float*)d_in[2];
    const float* Wkv  = (const float*)d_in[3];
    const float* Wo   = (const float*)d_in[4];
    const float* bo   = (const float*)d_in[5];
    float* out = (float*)d_out;

    unsigned short* ws = (unsigned short*)d_ws;
    const size_t SZ = (size_t)kM * kINNER;
    unsigned short* Qb = ws;
    unsigned short* Kb = ws + SZ;
    unsigned short* Vb = ws + 2 * SZ;
    unsigned short* Ob = ws + 3 * SZ;

    qkv_rope_kernel<<<dim3(kNQKV / 128, kM / 128), 256, 0, stream>>>(
        x, rope, Wq, Wkv, Qb, Kb, Vb);
    attn_kernel<<<dim3(kN / 128, kB * kH), 256, 0, stream>>>(Qb, Kb, Vb, Ob);
    oproj_kernel<<<dim3(kDIM / 128, kM / 128), 256, 0, stream>>>(Ob, Wo, bo, out);
}
